// Member_Aggregator_27230092657094
// MI455X (gfx1250) — matrix operand order repaired and checked by reading, not compiled, not run on hardware
//
#include <hip/hip_runtime.h>

#define N_NODES   40000
#define DEGREE    32
#define EMBED_DIM 64
#define NPB       8            // nodes (waves) per block

typedef __attribute__((ext_vector_type(16))) _Float16 v16h;
typedef __attribute__((ext_vector_type(8)))  _Float16 v8h;
typedef __attribute__((ext_vector_type(8)))  float    v8f;

__device__ __forceinline__ v8h cvt8_f32_f16(const float* __restrict__ p) {
  v8h r;
#pragma unroll
  for (int i = 0; i < 8; ++i) r[i] = (_Float16)p[i];
  return r;
}

__device__ __forceinline__ v8f wmma16(v16h a, v16h b, v8f c) {
  // D = A(16x32 f16) * B(32x16 f16) + C(16x16 f32)
  return __builtin_amdgcn_wmma_f32_16x16x32_f16(false, a, false, b, (short)0, c,
                                                false, false);
}

__global__ __launch_bounds__(256)
void member_agg_kernel(const int*   __restrict__ nodes,
                       const int*   __restrict__ neigh_idx,
                       const float* __restrict__ u2e,
                       const float* __restrict__ g2e,
                       const float* __restrict__ w1,
                       const float* __restrict__ b1,
                       const float* __restrict__ w2,
                       const float* __restrict__ b2,
                       const float* __restrict__ w3,
                       const float* __restrict__ b3,
                       float* __restrict__ out)
{
  __shared__ __attribute__((aligned(32))) _Float16 sW1[64 * 128]; // 16 KB
  __shared__ __attribute__((aligned(32))) _Float16 sW2[64 * 64];  //  8 KB
  __shared__ float sW3[64];
  __shared__ float sB1[64];
  __shared__ float sB2[64];
  __shared__ __attribute__((aligned(32))) _Float16 sH[NPB][DEGREE * 64]; // 32 KB

  const int tid = threadIdx.x;

  // ---- stage weights (f32 -> f16) into LDS, shared by all 8 waves ----
  for (int i = tid; i < 64 * 128; i += 256) sW1[i] = (_Float16)w1[i];
  for (int i = tid; i < 64 * 64;  i += 256) sW2[i] = (_Float16)w2[i];
  if (tid < 64) { sW3[tid] = w3[tid]; sB1[tid] = b1[tid]; sB2[tid] = b2[tid]; }
  __syncthreads();

  const int wave = tid >> 5;
  const int lane = tid & 31;
  const int lc   = lane & 15;   // column / row-within-tile index
  const int hh   = lane >> 4;   // which K-half this lane holds
  const int node = blockIdx.x * NPB + wave;
  if (node >= N_NODES) return;

  const int   g    = nodes[node];
  const float* grow = g2e + (long)g * EMBED_DIM;
  // lane L caches the user index of edge-row L of this node
  const int uid = neigh_idx[node * DEGREE + lane];
  const float b3v = b3[0];

  _Float16* __restrict__ H = &sH[wave][0];

  // =================== layer 1: X[32,128] @ W1^T -> H1[32,64] ===================
  v8f acc[2][4];
#pragma unroll
  for (int m = 0; m < 2; ++m)
#pragma unroll
    for (int n = 0; n < 4; ++n) { v8f z = {}; acc[m][n] = z; }

#pragma unroll
  for (int ks = 0; ks < 4; ++ks) {
    const int kbase = ks * 32;
    v16h afr[2];
#pragma unroll
    for (int m = 0; m < 2; ++m) {
      const int row  = m * 16 + lc;                 // edge row of this A lane
      const int ruid = __shfl(uid, row, 32);
      // cols [0,64) come from the gathered user row, [64,128) from the group row
      const float* src = (kbase < 64) ? (u2e + (long)ruid * EMBED_DIM + kbase)
                                      : (grow + (kbase - 64));
      union { v16h v; v8h h[2]; } u;
      u.h[0] = cvt8_f32_f16(src + 16 * hh);         // K = kbase + {0..7}/{16..23}
      u.h[1] = cvt8_f32_f16(src + 16 * hh + 8);     // K = kbase + {8..15}/{24..31}
      afr[m] = u.v;
    }
#pragma unroll
    for (int n = 0; n < 4; ++n) {
      const v16h bfr = *(const v16h*)&sW1[(n * 16 + lc) * 128 + kbase + 16 * hh];
      acc[0][n] = wmma16(afr[0], bfr, acc[0][n]);
      acc[1][n] = wmma16(afr[1], bfr, acc[1][n]);
    }
  }

  // bias + ReLU, transpose C-layout -> row-major f16 tile in LDS
#pragma unroll
  for (int m = 0; m < 2; ++m)
#pragma unroll
    for (int n = 0; n < 4; ++n) {
      const float bv = sB1[n * 16 + lc];
#pragma unroll
      for (int j = 0; j < 8; ++j) {
        float v = acc[m][n][j] + bv;
        v = v > 0.f ? v : 0.f;
        H[(m * 16 + j + 8 * hh) * 64 + n * 16 + lc] = (_Float16)v;
      }
    }

  // =================== layer 2: H1[32,64] @ W2^T -> H2[32,64] ===================
  v8f acc2[2][4];
#pragma unroll
  for (int m = 0; m < 2; ++m)
#pragma unroll
    for (int n = 0; n < 4; ++n) { v8f z = {}; acc2[m][n] = z; }

#pragma unroll
  for (int ks = 0; ks < 2; ++ks) {
    const int kbase = ks * 32;
    v16h afr[2];
#pragma unroll
    for (int m = 0; m < 2; ++m) {
      const int row = m * 16 + lc;
      union { v16h v; v8h h[2]; } u;
      u.h[0] = *(const v8h*)&H[row * 64 + kbase + 16 * hh];
      u.h[1] = *(const v8h*)&H[row * 64 + kbase + 16 * hh + 8];
      afr[m] = u.v;
    }
#pragma unroll
    for (int n = 0; n < 4; ++n) {
      const v16h bfr = *(const v16h*)&sW2[(n * 16 + lc) * 64 + kbase + 16 * hh];
      acc2[0][n] = wmma16(afr[0], bfr, acc2[0][n]);
      acc2[1][n] = wmma16(afr[1], bfr, acc2[1][n]);
    }
  }

  // bias + ReLU, store H2 (overwrites H1 tile; DS ops are in-order per wave)
#pragma unroll
  for (int m = 0; m < 2; ++m)
#pragma unroll
    for (int n = 0; n < 4; ++n) {
      const float bv = sB2[n * 16 + lc];
#pragma unroll
      for (int j = 0; j < 8; ++j) {
        float v = acc2[m][n][j] + bv;
        v = v > 0.f ? v : 0.f;
        H[(m * 16 + j + 8 * hh) * 64 + n * 16 + lc] = (_Float16)v;
      }
    }

  // =================== logits + segment softmax (one wave == one segment) =======
  float lg = b3v;
  {
    const _Float16* hrow = &H[lane * 64];
#pragma unroll
    for (int k = 0; k < 64; ++k) lg += (float)hrow[k] * sW3[k];
  }
  float mx = lg;
#pragma unroll
  for (int o = 16; o > 0; o >>= 1) mx = fmaxf(mx, __shfl_xor(mx, o, 32));
  const float ex = __expf(lg - mx);
  float sm = ex;
#pragma unroll
  for (int o = 16; o > 0; o >>= 1) sm += __shfl_xor(sm, o, 32);
  const float att = ex / sm;

  // =================== attention-weighted aggregation of member embeddings ======
  float o0 = 0.f, o1 = 0.f;   // lane covers output dims {lane, lane+32}
#pragma unroll 4
  for (int r = 0; r < DEGREE; ++r) {
    const float wgt = __shfl(att, r, 32);
    const int   ru  = __shfl(uid, r, 32);
    const float* pu = u2e + (long)ru * EMBED_DIM;
    o0 += wgt * pu[lane];
    o1 += wgt * pu[lane + 32];
  }
  out[(long)node * EMBED_DIM + lane]      = o0;
  out[(long)node * EMBED_DIM + lane + 32] = o1;
}

extern "C" void kernel_launch(void* const* d_in, const int* in_sizes, int n_in,
                              void* d_out, int out_size, void* d_ws, size_t ws_size,
                              hipStream_t stream) {
  const int*   nodes = (const int*)d_in[0];
  const int*   neigh = (const int*)d_in[1];
  // d_in[2] segment_ids unused: layout is implicit (DEGREE edges per node, sorted)
  const float* u2e   = (const float*)d_in[3];
  const float* g2e   = (const float*)d_in[4];
  const float* w1    = (const float*)d_in[5];
  const float* b1    = (const float*)d_in[6];
  const float* w2    = (const float*)d_in[7];
  const float* b2    = (const float*)d_in[8];
  const float* w3    = (const float*)d_in[9];
  const float* b3    = (const float*)d_in[10];

  dim3 grid((N_NODES + NPB - 1) / NPB);   // 5000 blocks x 256 threads (8 waves)
  member_agg_kernel<<<grid, 256, 0, stream>>>(nodes, neigh, u2e, g2e,
                                              w1, b1, w2, b2, w3, b3,
                                              (float*)d_out);
}
